// Block_36644660969860
// MI455X (gfx1250) — matrix operand order repaired and checked by reading, not compiled, not run on hardware
//
#include <hip/hip_runtime.h>
#include <hip/hip_bf16.h>

typedef __attribute__((ext_vector_type(16))) _Float16 v16h;
typedef __attribute__((ext_vector_type(8)))  float    v8f;

#define D_MODEL 384
#define SEQ_L   2048
#define BATCH   4
#define NHEAD   8
#define HDIM    48
#define QKV_OUT 1536
#define CATW    768
#define ROWS    (BATCH * SEQ_L)   // 8192

// ---------------- WMMA operand layout helpers (CDNA5 wave32) ----------------
// 16-bit A matrix 16x32: lane m=lane&15 holds row m; per-lane half i maps to
// K = (i&7) + 16*(i>=8) + 8*(lane>=16)   [ISA 7.12.2]
__device__ __forceinline__ int a_kmap(int i, int hi) {
    return (i & 7) + ((i >> 3) << 4) + (hi << 3);
}
// 16-bit B matrix 32x16: lane n=lane&15 holds column n; per-lane half i maps to
// K = (i&7) + 16*(i>=8) + 8*(lane>=16)   (same as A)
__device__ __forceinline__ int b_kmap(int i, int hi) {
    return (i & 7) + ((i >> 3) << 4) + (hi << 3);
}

__device__ __forceinline__ v8f wmma16(v16h a, v16h b, v8f c) {
    // D = A(16x32,f16) * B(32x16,f16) + C(16x16,f32)
    return __builtin_amdgcn_wmma_f32_16x16x32_f16(false, a, false, b,
                                                  (short)0, c, false, false);
}

// Load A operand from row-major src (row stride ld), rows m0..m0+15, K offset koff
__device__ __forceinline__ v16h load_A(const _Float16* __restrict__ base, int ld,
                                       int m0, int lane, int koff) {
    int m  = m0 + (lane & 15);
    int hi = lane >> 4;
    const _Float16* p = base + (long)m * ld + koff;
    v16h a;
#pragma unroll
    for (int i = 0; i < 16; ++i) a[i] = p[a_kmap(i, hi)];
    return a;
}

// Load B operand where B = W^T for row-major W[N][K] (column n of B = row n of W)
__device__ __forceinline__ v16h load_Bw(const _Float16* __restrict__ w, int ldw,
                                        int n0, int lane, int koff) {
    int n  = n0 + (lane & 15);
    int hi = lane >> 4;
    const _Float16* p = w + (long)n * ldw + koff;
    v16h b;
#pragma unroll
    for (int i = 0; i < 16; ++i) b[i] = p[b_kmap(i, hi)];
    return b;
}

// ---------------------------- small prep kernels ----------------------------
__global__ void cast_f32_f16_kernel(const float* __restrict__ src,
                                    _Float16* __restrict__ dst, int n) {
    int i = blockIdx.x * blockDim.x + threadIdx.x;
    if (i < n) dst[i] = (_Float16)src[i];
}

// BatchNorm statistics over (B,L) per channel: one block per channel.
__global__ void bn_stats_kernel(const float* __restrict__ x,
                                float* __restrict__ mean,
                                float* __restrict__ rstd) {
    int c = blockIdx.x;
    float s = 0.f, s2 = 0.f;
    for (int i = threadIdx.x; i < ROWS; i += blockDim.x) {
        float v = x[(long)i * D_MODEL + c];
        s += v; s2 += v * v;
    }
    __shared__ float sh[256], sh2[256];
    sh[threadIdx.x] = s; sh2[threadIdx.x] = s2;
    __syncthreads();
    for (int o = 128; o > 0; o >>= 1) {
        if ((int)threadIdx.x < o) {
            sh[threadIdx.x]  += sh[threadIdx.x + o];
            sh2[threadIdx.x] += sh2[threadIdx.x + o];
        }
        __syncthreads();
    }
    if (threadIdx.x == 0) {
        float m = sh[0] / (float)ROWS;
        float v = sh2[0] / (float)ROWS - m * m;   // biased variance
        mean[c] = m;
        rstd[c] = rsqrtf(v + 1e-5f);
    }
}

__global__ void bn_apply_kernel(const float* __restrict__ x,
                                const float* __restrict__ mean,
                                const float* __restrict__ rstd,
                                const float* __restrict__ gamma,
                                const float* __restrict__ beta,
                                _Float16* __restrict__ h16) {
    long i = (long)blockIdx.x * blockDim.x + threadIdx.x;
    if (i >= (long)ROWS * D_MODEL) return;
    int c = (int)(i % D_MODEL);
    h16[i] = (_Float16)((x[i] - mean[c]) * rstd[c] * gamma[c] + beta[c]);
}

// GELU(pre) -> cat[:, 0:384]
__global__ void gelu_kernel(const _Float16* __restrict__ proj,
                            _Float16* __restrict__ cat16) {
    long i = (long)blockIdx.x * blockDim.x + threadIdx.x;
    if (i >= (long)ROWS * D_MODEL) return;
    long row = i / D_MODEL;
    int  c   = (int)(i % D_MODEL);
    float p = (float)proj[row * QKV_OUT + 3 * D_MODEL + c];
    float g = 0.5f * p * (1.f + tanhf(0.7978845608028654f * (p + 0.044715f * p * p * p)));
    cat16[row * CATW + c] = (_Float16)g;
}

// --------------------------- QKV GEMM (WMMA) --------------------------------
// proj[8192,1536] = h[8192,384] * qkv_w^T + qkv_b
// Per-wave tile 32x64: 2 A operands share 4 B operands -> 8 WMMAs per K-step.
__global__ void gemm_qkv_kernel(const _Float16* __restrict__ h16,
                                const _Float16* __restrict__ w16,
                                const float* __restrict__ bias,
                                _Float16* __restrict__ proj) {
    int wid  = blockIdx.x * (blockDim.x >> 5) + (threadIdx.x >> 5);
    int lane = threadIdx.x & 31;
    const int MT = ROWS / 32;                 // 256
    int mt = wid % MT;
    int ng = wid / MT;                        // 0..23 (64-wide N groups)
    int m0 = mt * 32, n0 = ng * 64;
    v8f acc0[4] = {}, acc1[4] = {};
    for (int kk = 0; kk < D_MODEL; kk += 32) {
        v16h a0 = load_A(h16, D_MODEL, m0,      lane, kk);
        v16h a1 = load_A(h16, D_MODEL, m0 + 16, lane, kk);
#pragma unroll
        for (int t = 0; t < 4; ++t) {
            v16h b = load_Bw(w16, D_MODEL, n0 + 16 * t, lane, kk);
            acc0[t] = wmma16(a0, b, acc0[t]);
            acc1[t] = wmma16(a1, b, acc1[t]);
        }
    }
    int hi = lane >> 4, n = lane & 15;
#pragma unroll
    for (int t = 0; t < 4; ++t) {
        int col = n0 + 16 * t + n;
        float bv = bias[col];
#pragma unroll
        for (int r = 0; r < 8; ++r) {
            long row0 = m0 + r + (hi << 3);
            proj[row0 * QKV_OUT + col]        = (_Float16)(acc0[t][r] + bv);
            proj[(row0 + 16) * QKV_OUT + col] = (_Float16)(acc1[t][r] + bv);
        }
    }
}

// ---------------------- causal flash attention (WMMA) -----------------------
// One wave handles a 16-row Q tile; streams 32-key tiles; hd=48 padded to 64.
__global__ void attn_kernel(const _Float16* __restrict__ proj,
                            _Float16* __restrict__ cat16) {
    __shared__ _Float16 plds[4][16 * 32];
    int wid  = threadIdx.x >> 5;
    int lane = threadIdx.x & 31;
    int qt = blockIdx.x * 4 + wid;            // 0..127
    int h  = blockIdx.y;
    int b  = blockIdx.z;
    long rowbase = (long)b * SEQ_L;
    int qb = qt * 16;
    int n = lane & 15, hi = lane >> 4;

    const _Float16* qp = proj + h * HDIM;
    const _Float16* kp = proj + D_MODEL + h * HDIM;
    const _Float16* vp = proj + 2 * D_MODEL + h * HDIM;

    // Q as two A operands covering hd 0..31 and 32..47 (zero-padded to 64)
    v16h aq0, aq1;
    {
        const _Float16* p = qp + (rowbase + qb + n) * QKV_OUT;
#pragma unroll
        for (int i = 0; i < 16; ++i) {
            int k = a_kmap(i, hi);
            aq0[i] = p[k];
            aq1[i] = (k < 16) ? p[32 + k] : (_Float16)0.f;
        }
    }

    float Mrow[8], Srow[8];
#pragma unroll
    for (int r = 0; r < 8; ++r) { Mrow[r] = -1e30f; Srow[r] = 0.f; }
    v8f acc[3] = {};
    const float scale = 0.14433756729740643f;   // 1/sqrt(48)

    int jtiles = qt / 2 + 1;                    // keys needed = qb+16, tiles of 32
    for (int jt = 0; jt < jtiles; ++jt) {
        int jb = jt * 32;
        // Prefetch next tile's K rows into cache (global_prefetch_b8)
        if (jt + 1 < jtiles) {
            __builtin_prefetch(kp + (rowbase + jb + 32 + n) * QKV_OUT, 0, 3);
            __builtin_prefetch(kp + (rowbase + jb + 48 + n) * QKV_OUT, 0, 3);
            __builtin_prefetch(vp + (rowbase + jb + 32 + n) * QKV_OUT, 0, 3);
            __builtin_prefetch(vp + (rowbase + jb + 48 + n) * QKV_OUT, 0, 3);
        }
        // ---- S = Q K^T for two 16-key halves ----
        v8f s[2] = {};
#pragma unroll
        for (int nh = 0; nh < 2; ++nh) {
            const _Float16* p = kp + (rowbase + jb + nh * 16 + n) * QKV_OUT;
            v16h b0, b1;
#pragma unroll
            for (int i = 0; i < 16; ++i) {
                int k = b_kmap(i, hi);
                b0[i] = p[k];
                b1[i] = (k < 16) ? p[32 + k] : (_Float16)0.f;
            }
            s[nh] = wmma16(aq0, b0, s[nh]);
            s[nh] = wmma16(aq1, b1, s[nh]);
        }
        // ---- mask + scale + online softmax ----
        float tmax[8];
#pragma unroll
        for (int r = 0; r < 8; ++r) {
            int row = qb + r + (hi << 3);
            int c0 = jb + n, c1 = c0 + 16;
            float v0 = (c0 <= row) ? s[0][r] * scale : -1e30f;
            float v1 = (c1 <= row) ? s[1][r] * scale : -1e30f;
            s[0][r] = v0; s[1][r] = v1;
            tmax[r] = fmaxf(v0, v1);
        }
#pragma unroll
        for (int off = 1; off < 16; off <<= 1)
#pragma unroll
            for (int r = 0; r < 8; ++r)
                tmax[r] = fmaxf(tmax[r], __shfl_xor(tmax[r], off, 32));
        float alpha[8], psum[8];
#pragma unroll
        for (int r = 0; r < 8; ++r) {
            float mnew = fmaxf(Mrow[r], tmax[r]);
            alpha[r] = __expf(Mrow[r] - mnew);
            Mrow[r] = mnew;
            float p0 = __expf(s[0][r] - mnew);
            float p1 = __expf(s[1][r] - mnew);
            s[0][r] = p0; s[1][r] = p1;
            psum[r] = p0 + p1;
        }
#pragma unroll
        for (int off = 1; off < 16; off <<= 1)
#pragma unroll
            for (int r = 0; r < 8; ++r)
                psum[r] += __shfl_xor(psum[r], off, 32);
#pragma unroll
        for (int r = 0; r < 8; ++r) Srow[r] = Srow[r] * alpha[r] + psum[r];
#pragma unroll
        for (int t = 0; t < 3; ++t)
#pragma unroll
            for (int r = 0; r < 8; ++r) acc[t][r] *= alpha[r];

        // ---- stage P (16x32 f16) through LDS to re-layout C->A ----
        _Float16* pl = plds[wid];
#pragma unroll
        for (int r = 0; r < 8; ++r) {
            int row = r + (hi << 3);
            pl[row * 32 + n]      = (_Float16)s[0][r];
            pl[row * 32 + 16 + n] = (_Float16)s[1][r];
        }
        asm volatile("s_wait_dscnt 0" ::: "memory");  // LDS per-wave in-order; explicit drain
        v16h ap;
#pragma unroll
        for (int i = 0; i < 16; ++i) ap[i] = pl[n * 32 + a_kmap(i, hi)];

        // ---- O += P * V  (3 N-tiles cover hd=48) ----
#pragma unroll
        for (int t = 0; t < 3; ++t) {
            v16h bv;
#pragma unroll
            for (int i = 0; i < 16; ++i) {
                int k = b_kmap(i, hi);
                bv[i] = vp[(rowbase + jb + k) * QKV_OUT + t * 16 + n];
            }
            acc[t] = wmma16(ap, bv, acc[t]);
        }
    }

    // ---- normalize and write into cat[:, 384 + h*48 + d] ----
#pragma unroll
    for (int r = 0; r < 8; ++r) {
        float inv = 1.0f / Srow[r];
        long row = rowbase + qb + r + (hi << 3);
#pragma unroll
        for (int t = 0; t < 3; ++t)
            cat16[row * CATW + D_MODEL + h * HDIM + t * 16 + n] =
                (_Float16)(acc[t][r] * inv);
    }
}

// ------------------------ output GEMM + residual ----------------------------
// out[8192,384] = x + cat[8192,768] * out_w^T + out_b ; per-wave tile 32x64
__global__ void gemm_out_kernel(const _Float16* __restrict__ cat16,
                                const _Float16* __restrict__ w16,
                                const float* __restrict__ bias,
                                const float* __restrict__ x,
                                float* __restrict__ out) {
    int wid  = blockIdx.x * (blockDim.x >> 5) + (threadIdx.x >> 5);
    int lane = threadIdx.x & 31;
    const int MT = ROWS / 32;                 // 256
    int mt = wid % MT;
    int ng = wid / MT;                        // 0..5
    int m0 = mt * 32, n0 = ng * 64;
    v8f acc0[4] = {}, acc1[4] = {};
    for (int kk = 0; kk < CATW; kk += 32) {
        v16h a0 = load_A(cat16, CATW, m0,      lane, kk);
        v16h a1 = load_A(cat16, CATW, m0 + 16, lane, kk);
#pragma unroll
        for (int t = 0; t < 4; ++t) {
            v16h b = load_Bw(w16, CATW, n0 + 16 * t, lane, kk);
            acc0[t] = wmma16(a0, b, acc0[t]);
            acc1[t] = wmma16(a1, b, acc1[t]);
        }
    }
    int hi = lane >> 4, n = lane & 15;
#pragma unroll
    for (int t = 0; t < 4; ++t) {
        int col = n0 + 16 * t + n;
        float bv = bias[col];
#pragma unroll
        for (int r = 0; r < 8; ++r) {
            long row0 = m0 + r + (hi << 3);
            out[row0 * D_MODEL + col] = x[row0 * D_MODEL + col] + acc0[t][r] + bv;
            out[(row0 + 16) * D_MODEL + col] =
                x[(row0 + 16) * D_MODEL + col] + acc1[t][r] + bv;
        }
    }
}

// ------------------------------- launcher -----------------------------------
extern "C" void kernel_launch(void* const* d_in, const int* in_sizes, int n_in,
                              void* d_out, int out_size, void* d_ws, size_t ws_size,
                              hipStream_t stream) {
    const float* x    = (const float*)d_in[0];
    const float* gmm  = (const float*)d_in[1];
    const float* beta = (const float*)d_in[2];
    const float* qkvw = (const float*)d_in[3];
    const float* qkvb = (const float*)d_in[4];
    const float* outw = (const float*)d_in[5];
    const float* outb = (const float*)d_in[6];
    float* out = (float*)d_out;

    char* ws = (char*)d_ws;
    size_t off = 0;
    auto walloc = [&](size_t bytes) -> void* {
        void* p = ws + off;
        off = (off + bytes + 255) & ~(size_t)255;
        return p;
    };
    _Float16* qkvw16 = (_Float16*)walloc((size_t)QKV_OUT * D_MODEL * 2);
    _Float16* outw16 = (_Float16*)walloc((size_t)D_MODEL * CATW * 2);
    float*    mean   = (float*)walloc(D_MODEL * 4);
    float*    rstd   = (float*)walloc(D_MODEL * 4);
    _Float16* h16    = (_Float16*)walloc((size_t)ROWS * D_MODEL * 2);
    _Float16* proj16 = (_Float16*)walloc((size_t)ROWS * QKV_OUT * 2);
    _Float16* cat16  = (_Float16*)walloc((size_t)ROWS * CATW * 2);

    cast_f32_f16_kernel<<<(QKV_OUT * D_MODEL + 255) / 256, 256, 0, stream>>>(
        qkvw, qkvw16, QKV_OUT * D_MODEL);
    cast_f32_f16_kernel<<<(D_MODEL * CATW + 255) / 256, 256, 0, stream>>>(
        outw, outw16, D_MODEL * CATW);
    bn_stats_kernel<<<D_MODEL, 256, 0, stream>>>(x, mean, rstd);
    bn_apply_kernel<<<(ROWS * D_MODEL + 255) / 256, 256, 0, stream>>>(
        x, mean, rstd, gmm, beta, h16);
    // 256 M-tiles x 24 N-groups = 6144 waves / 4 per block
    gemm_qkv_kernel<<<(256 * 24) / 4, 128, 0, stream>>>(h16, qkvw16, qkvb, proj16);
    gelu_kernel<<<(ROWS * D_MODEL + 255) / 256, 256, 0, stream>>>(proj16, cat16);
    attn_kernel<<<dim3(32, NHEAD, BATCH), 128, 0, stream>>>(proj16, cat16);
    // 256 M-tiles x 6 N-groups = 1536 waves / 4 per block
    gemm_out_kernel<<<(256 * 6) / 4, 128, 0, stream>>>(cat16, outw16, outb, x, out);
}
